// SigTKANCell_7765300871695
// MI455X (gfx1250) — matrix operand order repaired and checked by reading, not compiled, not run on hardware
//
#include <hip/hip_runtime.h>

// ---------------------------------------------------------------------------
// CDNA5 (gfx1250, wave32) implementation of SigTKANCell forward.
// Dominant GEMMs run on v_wmma_f32_16x16x32_bf16. Spline/silu activations are
// generated on-the-fly into LDS (never materialized in HBM); weights are
// converted once per call into WMMA-fragment order in workspace.
// ---------------------------------------------------------------------------

typedef float  v8f   __attribute__((ext_vector_type(8)));
typedef __bf16 v16bf __attribute__((ext_vector_type(16)));
typedef __bf16 v8bf  __attribute__((ext_vector_type(8)));

union BF16x16 { v16bf v; v8bf h[2]; };

__device__ __forceinline__ float sigmoidf_(float x) { return 1.0f / (1.0f + __expf(-x)); }
__device__ __forceinline__ float siluf_(float x)    { return x * sigmoidf_(x); }

__device__ __forceinline__ float wave_sum(float v) {
#pragma unroll
  for (int m = 16; m > 0; m >>= 1) v += __shfl_xor(v, m, 32);
  return v;
}

// Cox-de Boor, uniform knots t_i = -1 + (i-3)*0.4, i=0..11 -> 8 cubic bases.
__device__ __forceinline__ void bspline8(float x, float* b3) {
  const float INV1 = 2.5f, INV2 = 1.25f, INV3 = (1.0f / 1.2f);
  float b0[11], b1[10], b2[9];
#pragma unroll
  for (int t = 0; t < 11; ++t) {
    float g0 = -1.0f + (t - 3) * 0.4f;
    b0[t] = (x >= g0 && x < g0 + 0.4f) ? 1.0f : 0.0f;
  }
#pragma unroll
  for (int t = 0; t < 10; ++t) {
    float g = -1.0f + (t - 3) * 0.4f;
    b1[t] = (x - g) * INV1 * b0[t] + ((g + 0.8f) - x) * INV1 * b0[t + 1];
  }
#pragma unroll
  for (int t = 0; t < 9; ++t) {
    float g = -1.0f + (t - 3) * 0.4f;
    b2[t] = (x - g) * INV2 * b1[t] + ((g + 1.2f) - x) * INV2 * b1[t + 1];
  }
#pragma unroll
  for (int t = 0; t < 8; ++t) {
    float g = -1.0f + (t - 3) * 0.4f;
    b3[t] = (x - g) * INV3 * b2[t] + ((g + 1.6f) - x) * INV3 * b2[t + 1];
  }
}

// ---------------------------------------------------------------------------
__global__ void k_zero(float* p, int n) {
  int i = blockIdx.x * blockDim.x + threadIdx.x;
  if (i < n) p[i] = 0.0f;
}

// Analytic LayerNorm stats of sig = [x, 0.5*x(x)x] from Sx, Sx2 (len 16512).
__global__ void k_stats_sig(const float* __restrict__ x, float2* __restrict__ st) {
  int wid = threadIdx.x >> 5, lane = threadIdx.x & 31;
  int row = blockIdx.x * 4 + wid;
  float s = 0.f, s2 = 0.f;
#pragma unroll
  for (int t = 0; t < 4; ++t) { float v = x[row * 128 + lane + t * 32]; s += v; s2 += v * v; }
  s = wave_sum(s); s2 = wave_sum(s2);
  if (lane == 0) {
    float sumsig = s + 0.5f * s * s;          // sum over outer = 0.5*(Sx)^2
    float sumsq  = s2 + 0.25f * s2 * s2;      // sumsq over outer = 0.25*(Sx2)^2
    const float invn = 1.0f / 16512.0f;
    float mu  = sumsig * invn;
    float var = sumsq * invn - mu * mu;
    st[row] = make_float2(mu, rsqrtf(var + 1e-3f));
  }
}

// Per-row mean/var for K-wide rows (one wave per row).
__global__ void k_stats_rows(const float* __restrict__ X, float2* __restrict__ st, int K, int rows) {
  int wid = threadIdx.x >> 5, lane = threadIdx.x & 31;
  int row = blockIdx.x * (blockDim.x >> 5) + wid;
  if (row >= rows) return;
  float s = 0.f, s2 = 0.f;
  for (int k = lane; k < K; k += 32) { float v = X[(size_t)row * K + k]; s += v; s2 += v * v; }
  s = wave_sum(s); s2 = wave_sum(s2);
  if (!lane) {
    float mu = s / K, var = s2 / K - mu * mu;
    st[row] = make_float2(mu, rsqrtf(var + 1e-3f));
  }
}

// ---------------------------------------------------------------------------
// B operand (KxN, row-major f32) -> bf16 WMMA fragment order.
// Fragment layout: lane L holds col = n0 + (L&15); L<16 holds kk {0..7,16..23}, L>=16 {8..15,24..31}.
__global__ void k_pack_B2(const float* __restrict__ A, int Ka, int lda,
                          const float* __restrict__ Bsrc, int ldb,
                          __bf16* __restrict__ dst, int N, int total) {
  int t = blockIdx.x * blockDim.x + threadIdx.x;
  if (t >= total) return;
  int lane = t & 31, rest = t >> 5;
  int nch = rest % (N >> 4), kstep = rest / (N >> 4);
  int col = nch * 16 + (lane & 15);
  int kbase = kstep * 32 + (lane >> 4) * 8;
  __bf16* d = dst + (size_t)t * 16;
#pragma unroll
  for (int e = 0; e < 16; ++e) {
    int k = kbase + e + (e >= 8 ? 8 : 0);
    float v = (k < Ka) ? A[(size_t)k * lda + col] : Bsrc[(size_t)(k - Ka) * ldb + col];
    d[e] = (__bf16)v;
  }
}

// A operand (MxK, two concatenated row-major f32 sources) -> bf16 fragments.
// lane L: row = mchunk*16 + (L&15); L<16 holds kk {0..7,16..23}, L>=16 {8..15,24..31}.
__global__ void k_pack_A2(const float* __restrict__ A, int Ka, int lda,
                          const float* __restrict__ Bsrc, int ldb,
                          __bf16* __restrict__ dst, int Ksteps, int total) {
  int t = blockIdx.x * blockDim.x + threadIdx.x;
  if (t >= total) return;
  int lane = t & 31, rest = t >> 5;
  int kstep = rest % Ksteps, mchunk = rest / Ksteps;
  int row = mchunk * 16 + (lane & 15);
  int half = lane >> 4;
  __bf16* d = dst + (size_t)t * 16;
#pragma unroll
  for (int e = 0; e < 16; ++e) {
    int kk = (e < 8) ? (half * 8 + e) : (half * 8 + e + 8);
    int k = kstep * 32 + kk;
    float v = (k < Ka) ? A[(size_t)row * lda + k] : Bsrc[(size_t)row * ldb + (k - Ka)];
    d[e] = (__bf16)v;
  }
}

// ---------------------------------------------------------------------------
// Generic WMMA GEMM over pre-packed fragments. BM=128, BN=128, 8 waves.
__global__ __launch_bounds__(256) void k_gemm_frag(const __bf16* __restrict__ A,
                                                   const __bf16* __restrict__ B,
                                                   float* __restrict__ C, int Ksteps, int Nld) {
  int lane = threadIdx.x & 31, wid = threadIdx.x >> 5;
  int waveM = wid >> 1, waveN = wid & 1;
  int nchunks = Nld >> 4;
  v8f zero = {};
  v8f acc[2][4];
#pragma unroll
  for (int a = 0; a < 2; ++a)
#pragma unroll
    for (int b = 0; b < 4; ++b) acc[a][b] = zero;

  for (int ks = 0; ks < Ksteps; ++ks) {
    v16bf wf[4];
#pragma unroll
    for (int c = 0; c < 4; ++c) {
      int nch = blockIdx.y * 8 + waveN * 4 + c;
      wf[c] = *(const v16bf*)(B + ((size_t)(ks * nchunks + nch) * 32 + lane) * 16);
    }
#pragma unroll
    for (int mc = 0; mc < 2; ++mc) {
      int mchunk = blockIdx.x * 8 + waveM * 2 + mc;
      v16bf af = *(const v16bf*)(A + ((size_t)(mchunk * Ksteps + ks) * 32 + lane) * 16);
#pragma unroll
      for (int c = 0; c < 4; ++c)
        acc[mc][c] = __builtin_amdgcn_wmma_f32_16x16x32_bf16(false, af, false, wf[c],
                                                             (short)0, acc[mc][c], false, false);
    }
  }
  int rbase = blockIdx.x * 128 + waveM * 32 + (lane >> 4) * 8;
  int cbase = blockIdx.y * 128 + waveN * 64 + (lane & 15);
#pragma unroll
  for (int mc = 0; mc < 2; ++mc)
#pragma unroll
    for (int c = 0; c < 4; ++c)
#pragma unroll
      for (int i = 0; i < 8; ++i)
        C[(size_t)(rbase + mc * 16 + i) * Nld + cbase + c * 16] = acc[mc][c][i];
}

// ---------------------------------------------------------------------------
// Fused KANLinear GEMM: out += silu(LN(x)) @ Wb + bsplines(LN(x)) @ Ws.
// SIG mode: feature j of row b is sig[b,j] computed analytically from inputs.
// Activations for a 32-feature block are generated into LDS once per WG and
// consumed by all waves as WMMA A-fragments; weights stream as fragments.
template <bool SIG, int NWN>
__global__ __launch_bounds__(1024) void k_kan_gemm(
    const float* __restrict__ X, int ldx, const float2* __restrict__ stats,
    const float* __restrict__ lng, const float* __restrict__ lnb,
    const __bf16* __restrict__ Wb, const __bf16* __restrict__ Ws,
    float* __restrict__ out, int Kfeat, int fbPerChunk, int Nld,
    long xz, long statz, long lnz, long wbz, long wsz, long outz) {
  int z = blockIdx.z;
  X += (size_t)z * xz; stats += (size_t)z * statz;
  lng += (size_t)z * lnz; lnb += (size_t)z * lnz;
  Wb += (size_t)z * wbz; Ws += (size_t)z * wsz; out += (size_t)z * outz;

  __shared__ alignas(16) __bf16 sA1[128][40];   // silu tile, padded rows
  __shared__ alignas(16) __bf16 sA2[128][264];  // 8-basis tile, padded rows
  __shared__ float2 sStats[128];

  const int tid = threadIdx.x;
  const int lane = tid & 31, wid = tid >> 5;
  const int m0 = blockIdx.x * 128;
  const int totalFb = Kfeat >> 5;
  int fb0 = blockIdx.y * fbPerChunk;
  int fb1 = fb0 + fbPerChunk; if (fb1 > totalFb) fb1 = totalFb;

  if (tid < 128) sStats[tid] = stats[m0 + tid];

  const int waveM = wid / NWN, waveN = wid % NWN;
  const int nchunks = Nld >> 4;

  v8f zero = {};
  v8f acc[2][4];
#pragma unroll
  for (int a = 0; a < 2; ++a)
#pragma unroll
    for (int b = 0; b < 4; ++b) acc[a][b] = zero;

  for (int fb = fb0; fb < fb1; ++fb) {
    int j0 = fb << 5;
    __syncthreads();
    // --- generate silu + spline bases for 128 rows x 32 features ---
    for (int p = tid; p < 4096; p += blockDim.x) {
      int row = p >> 5, f = p & 31, j = j0 + f;
      float sv;
      if (SIG) {
        if (j < 128) sv = X[(size_t)(m0 + row) * ldx + j];
        else {
          int q = j - 128;
          sv = 0.5f * X[(size_t)(m0 + row) * ldx + (q >> 7)] *
                      X[(size_t)(m0 + row) * ldx + (q & 127)];
        }
      } else {
        sv = X[(size_t)(m0 + row) * ldx + j];
      }
      float2 st = sStats[row];
      float xn = (sv - st.x) * st.y * lng[j] + lnb[j];
      sA1[row][f] = (__bf16)siluf_(xn);
      float bs[8]; bspline8(xn, bs);
      v8bf pk;
#pragma unroll
      for (int r = 0; r < 8; ++r) pk[r] = (__bf16)bs[r];
      *(v8bf*)&sA2[row][f * 8] = pk;
    }
    __syncthreads();

    // --- base (silu) phase: one K-step ---
    {
      if (fb + 1 < fb1)
        __builtin_prefetch(Wb + ((size_t)((fb + 1) * nchunks + waveN * 4) * 32 + lane) * 16, 0, 1);
      v16bf wf[4];
#pragma unroll
      for (int c = 0; c < 4; ++c) {
        int nch = waveN * 4 + c;
        wf[c] = *(const v16bf*)(Wb + ((size_t)(fb * nchunks + nch) * 32 + lane) * 16);
      }
#pragma unroll
      for (int mc = 0; mc < 2; ++mc) {
        int rl = waveM * 32 + mc * 16 + (lane & 15);
        int kb = (lane >> 4) * 8;
        BF16x16 a;
        a.h[0] = *(const v8bf*)&sA1[rl][kb];
        a.h[1] = *(const v8bf*)&sA1[rl][kb + 16];
#pragma unroll
        for (int c = 0; c < 4; ++c)
          acc[mc][c] = __builtin_amdgcn_wmma_f32_16x16x32_bf16(false, a.v, false, wf[c],
                                                               (short)0, acc[mc][c], false, false);
      }
    }
    // --- spline phase: 8 K-steps ---
#pragma unroll
    for (int s = 0; s < 8; ++s) {
      v16bf wf[4];
#pragma unroll
      for (int c = 0; c < 4; ++c) {
        int nch = waveN * 4 + c;
        wf[c] = *(const v16bf*)(Ws + ((size_t)((fb * 8 + s) * nchunks + nch) * 32 + lane) * 16);
      }
#pragma unroll
      for (int mc = 0; mc < 2; ++mc) {
        int rl = waveM * 32 + mc * 16 + (lane & 15);
        int kb = (lane >> 4) * 8;
        BF16x16 a;
        a.h[0] = *(const v8bf*)&sA2[rl][s * 32 + kb];
        a.h[1] = *(const v8bf*)&sA2[rl][s * 32 + kb + 16];
#pragma unroll
        for (int c = 0; c < 4; ++c)
          acc[mc][c] = __builtin_amdgcn_wmma_f32_16x16x32_bf16(false, a.v, false, wf[c],
                                                               (short)0, acc[mc][c], false, false);
      }
    }
  }
  // K-split partial accumulation
  int rbase = m0 + waveM * 32 + (lane >> 4) * 8;
  int cbase = waveN * 64 + (lane & 15);
#pragma unroll
  for (int mc = 0; mc < 2; ++mc)
#pragma unroll
    for (int c = 0; c < 4; ++c)
#pragma unroll
      for (int i = 0; i < 8; ++i)
        atomicAdd(&out[(size_t)(rbase + mc * 16 + i) * Nld + cbase + c * 16], acc[mc][c][i]);
}

// ---------------------------------------------------------------------------
// h2 = kan_linear(h1): LN over 512, tiny N=3 output. One wave per row.
__global__ void k_h2(const float* __restrict__ h1, const float* __restrict__ lng,
                     const float* __restrict__ lnb, const float* __restrict__ wbase,
                     const float* __restrict__ wspline, float* __restrict__ h2) {
  int lane = threadIdx.x & 31, wid = threadIdx.x >> 5;
  int row = blockIdx.x * 8 + wid;
  const float* xr = h1 + (size_t)row * 512;
  float v[16], s = 0.f, s2 = 0.f;
#pragma unroll
  for (int t = 0; t < 16; ++t) { v[t] = xr[lane + t * 32]; s += v[t]; s2 += v[t] * v[t]; }
  s = wave_sum(s); s2 = wave_sum(s2);
  float mu = s * (1.f / 512.f), var = s2 * (1.f / 512.f) - mu * mu, rstd = rsqrtf(var + 1e-3f);
  float a0 = 0.f, a1 = 0.f, a2 = 0.f;
#pragma unroll
  for (int t = 0; t < 16; ++t) {
    int j = lane + t * 32;
    float xn = (v[t] - mu) * rstd * lng[j] + lnb[j];
    float sl = siluf_(xn);
    a0 += sl * wbase[j * 3 + 0]; a1 += sl * wbase[j * 3 + 1]; a2 += sl * wbase[j * 3 + 2];
    float bs[8]; bspline8(xn, bs);
#pragma unroll
    for (int r = 0; r < 8; ++r) {
      const float* wr = wspline + (size_t)(j * 8 + r) * 3;
      a0 += bs[r] * wr[0]; a1 += bs[r] * wr[1]; a2 += bs[r] * wr[2];
    }
  }
  a0 = wave_sum(a0); a1 = wave_sum(a1); a2 = wave_sum(a2);
  if (!lane) { h2[row * 3 + 0] = a0; h2[row * 3 + 1] = a1; h2[row * 3 + 2] = a2; }
}

// skip = sig @ g_skip_w + b (sig generated on the fly). One block per row.
__global__ void k_skip(const float* __restrict__ x, const float* __restrict__ w,
                       const float* __restrict__ b, float* __restrict__ skip) {
  __shared__ float xr[128];
  __shared__ float red[3][128];
  int tid = threadIdx.x, row = blockIdx.x;
  xr[tid] = x[(size_t)row * 128 + tid];
  __syncthreads();
  float a0 = 0.f, a1 = 0.f, a2 = 0.f;
  for (int j = tid; j < 16512; j += 128) {
    float sv = (j < 128) ? xr[j] : 0.5f * xr[(j - 128) >> 7] * xr[(j - 128) & 127];
    const float* wr = w + (size_t)j * 3;
    a0 += sv * wr[0]; a1 += sv * wr[1]; a2 += sv * wr[2];
  }
  red[0][tid] = a0; red[1][tid] = a1; red[2][tid] = a2;
  __syncthreads();
  for (int s = 64; s > 0; s >>= 1) {
    if (tid < s) {
      red[0][tid] += red[0][tid + s];
      red[1][tid] += red[1][tid + s];
      red[2][tid] += red[2][tid + s];
    }
    __syncthreads();
  }
  if (tid < 3) skip[row * 3 + tid] = red[tid][0] + b[tid];
}

// GLU gate + LN + softmax over n=3 -> signature weights.
__global__ void k_sigw(const float* __restrict__ h2, const float* __restrict__ skip,
                       const float* __restrict__ gw, const float* __restrict__ gb,
                       const float* __restrict__ sw, const float* __restrict__ sb,
                       const float* __restrict__ lg, const float* __restrict__ lb,
                       float* __restrict__ sigw) {
  int bi = blockIdx.x * blockDim.x + threadIdx.x;
  if (bi >= 1024) return;
  float h[3] = {h2[bi * 3], h2[bi * 3 + 1], h2[bi * 3 + 2]};
  float y[3];
#pragma unroll
  for (int n = 0; n < 3; ++n) {
    float g = gb[n], sg = sb[n];
#pragma unroll
    for (int k = 0; k < 3; ++k) { g += h[k] * gw[k * 3 + n]; sg += h[k] * sw[k * 3 + n]; }
    y[n] = skip[bi * 3 + n] + g * sigmoidf_(sg);
  }
  float mu = (y[0] + y[1] + y[2]) * (1.f / 3.f);
  float var = ((y[0] - mu) * (y[0] - mu) + (y[1] - mu) * (y[1] - mu) + (y[2] - mu) * (y[2] - mu)) * (1.f / 3.f);
  float rstd = rsqrtf(var + 1e-3f);
  float e[3], mx = -1e30f;
#pragma unroll
  for (int n = 0; n < 3; ++n) { e[n] = (y[n] - mu) * rstd * lg[n] + lb[n]; mx = fmaxf(mx, e[n]); }
  float se = 0.f;
#pragma unroll
  for (int n = 0; n < 3; ++n) { e[n] = __expf(e[n] - mx); se += e[n]; }
#pragma unroll
  for (int n = 0; n < 3; ++n) sigw[bi * 3 + n] = e[n] / se;
}

// Weight sub_out, update recurrent states (output #3), pack agg A-fragments.
__global__ void k_sub_final(const float* __restrict__ kan, const float* __restrict__ sigw,
                            const float* __restrict__ sub_kernel, const float* __restrict__ sub_states,
                            float* __restrict__ out_states, __bf16* __restrict__ Aagg) {
  int t = blockIdx.x * blockDim.x + threadIdx.x;  // < 3*1024*128
  int n = t >> 17, rem = t & 131071;
  int b = rem >> 7, o = rem & 127;
  float so = kan[t] * sigw[b * 3 + n];
  float rk_h = sub_kernel[n * 256 + o], rk_x = sub_kernel[n * 256 + 128 + o];
  out_states[t] = rk_h * so + sub_states[t] * rk_x;
  // agg[b, k=n*128+o] -> A-fragment slot (Ksteps = 12)
  int k = n * 128 + o;
  int kstep = k >> 5, kk = k & 31;
  int lane = (b & 15) + 16 * ((kk >> 3) & 1);
  int e = (kk & 7) + 8 * (kk >> 4);
  int mchunk = b >> 4;
  Aagg[((size_t)(mchunk * 12 + kstep) * 32 + lane) * 16 + e] = (__bf16)so;
}

// LSTM tail: gates sigmoid(+bias), c = f*c0 + i*cg, h = o*tanh(c).
__global__ void k_final(const float* __restrict__ zpre, const float* __restrict__ bias,
                        const float* __restrict__ xopre, const float* __restrict__ aggb,
                        const float* __restrict__ c0, float* __restrict__ out) {
  int t = blockIdx.x * blockDim.x + threadIdx.x;  // < 1024*512
  int b = t >> 9, u = t & 511;
  const float* zp = zpre + (size_t)b * 1536;
  float xi = sigmoidf_(zp[u] + bias[u]);
  float xf = sigmoidf_(zp[512 + u] + bias[512 + u]);
  float xc = sigmoidf_(zp[1024 + u] + bias[1024 + u]);
  float xo = sigmoidf_(xopre[t] + aggb[u]);
  float c = xf * c0[t] + xi * xc;
  out[t] = xo * tanhf(c);
  out[524288 + t] = c;
}

// ---------------------------------------------------------------------------
extern "C" void kernel_launch(void* const* d_in, const int* in_sizes, int n_in,
                              void* d_out, int out_size, void* d_ws, size_t ws_size,
                              hipStream_t stream) {
  const float* inputs        = (const float*)d_in[0];
  const float* h0            = (const float*)d_in[1];
  const float* c0            = (const float*)d_in[2];
  const float* sub_states    = (const float*)d_in[3];
  const float* kernelw       = (const float*)d_in[4];
  const float* recw          = (const float*)d_in[5];
  const float* bias          = (const float*)d_in[6];
  const float* sub_kernel    = (const float*)d_in[7];
  const float* sub_rk_inputs = (const float*)d_in[8];
  const float* sub_rk_states = (const float*)d_in[9];
  const float* agg_w         = (const float*)d_in[10];
  const float* agg_b         = (const float*)d_in[11];
  const float* sub_ln_g      = (const float*)d_in[12];
  const float* sub_ln_b      = (const float*)d_in[13];
  const float* sub_base_w    = (const float*)d_in[14];
  const float* sub_spline_w  = (const float*)d_in[15];
  const float* g_skip_w      = (const float*)d_in[16];
  const float* g_skip_b      = (const float*)d_in[17];
  const float* g_k1_ln_g     = (const float*)d_in[18];
  const float* g_k1_ln_b     = (const float*)d_in[19];
  const float* g_k1_base     = (const float*)d_in[20];
  const float* g_k1_spline   = (const float*)d_in[21];
  const float* g_k2_ln_g     = (const float*)d_in[22];
  const float* g_k2_ln_b     = (const float*)d_in[23];
  const float* g_k2_base     = (const float*)d_in[24];
  const float* g_k2_spline   = (const float*)d_in[25];
  const float* g_gate_w      = (const float*)d_in[26];
  const float* g_gate_b      = (const float*)d_in[27];
  const float* g_sig_w       = (const float*)d_in[28];
  const float* g_sig_b       = (const float*)d_in[29];
  const float* g_ln_g        = (const float*)d_in[30];
  const float* g_ln_b        = (const float*)d_in[31];
  (void)in_sizes; (void)n_in; (void)out_size; (void)ws_size;

  char* wsb = (char*)d_ws;
  size_t off = 0;
  auto alloc = [&](size_t bytes) -> char* {
    char* p = wsb + off;
    off = (off + bytes + 255) & ~(size_t)255;
    return p;
  };
  __bf16* WB1  = (__bf16*)alloc(8454144ull * 2);   // g_k1_base frags  (16512x512)
  __bf16* WS1  = (__bf16*)alloc(67633152ull * 2);  // g_k1_spline frags (132096x512)
  __bf16* BZ   = (__bf16*)alloc(983040ull * 2);    // [kernel;recurrent] (640x1536)
  __bf16* BSRK = (__bf16*)alloc(98304ull * 2);     // sub_rk combined (3 x 256x128)
  __bf16* BSB  = (__bf16*)alloc(49152ull * 2);     // sub_base (3 x 128x128)
  __bf16* BSS  = (__bf16*)alloc(393216ull * 2);    // sub_spline (3 x 1024x128)
  __bf16* BAGG = (__bf16*)alloc(196608ull * 2);    // agg_w (384x512)
  __bf16* AZ   = (__bf16*)alloc(655360ull * 2);    // [inputs|h0] (1024x640)
  __bf16* ASUB = (__bf16*)alloc(786432ull * 2);    // [inputs|sub_states] (3 x 1024x256)
  __bf16* AAGG = (__bf16*)alloc(393216ull * 2);    // agg (1024x384)
  float2* SIGST = (float2*)alloc(1024 * sizeof(float2));
  float2* SUBST = (float2*)alloc(3072 * sizeof(float2));
  float* H1    = (float*)alloc(524288ull * 4);
  float* H2v   = (float*)alloc(3072ull * 4);
  float* SKIP  = (float*)alloc(3072ull * 4);
  float* SIGW  = (float*)alloc(3072ull * 4);
  float* ZPRE  = (float*)alloc(1572864ull * 4);
  float* AGGIN = (float*)alloc(393216ull * 4);
  float* SUBKAN = (float*)alloc(393216ull * 4);
  float* XOPRE = (float*)alloc(524288ull * 4);

  // zero atomic-accumulated buffers (every call; ws is not re-poisoned)
  k_zero<<<2048, 256, 0, stream>>>(H1, 524288);
  k_zero<<<1536, 256, 0, stream>>>(SUBKAN, 393216);
  k_stats_sig<<<256, 128, 0, stream>>>(inputs, SIGST);

  // weight conversion + fragment swizzle (bf16)
  { int tot = 516 * 32 * 32;
    k_pack_B2<<<(tot + 255) / 256, 256, 0, stream>>>(g_k1_base, 16512, 512, (const float*)nullptr, 0, WB1, 512, tot); }
  { int tot = 4128 * 32 * 32;
    k_pack_B2<<<(tot + 255) / 256, 256, 0, stream>>>(g_k1_spline, 132096, 512, (const float*)nullptr, 0, WS1, 512, tot); }
  { int tot = 20 * 96 * 32;
    k_pack_B2<<<(tot + 255) / 256, 256, 0, stream>>>(kernelw, 128, 1536, recw, 1536, BZ, 1536, tot); }
  for (int n = 0; n < 3; ++n) {
    int tot = 8 * 8 * 32;
    k_pack_B2<<<(tot + 255) / 256, 256, 0, stream>>>(sub_rk_inputs + n * 16384, 128, 128,
                                                     sub_rk_states + n * 16384, 128,
                                                     BSRK + n * 32768, 128, tot);
    tot = 4 * 8 * 32;
    k_pack_B2<<<(tot + 255) / 256, 256, 0, stream>>>(sub_base_w + n * 16384, 128, 128,
                                                     (const float*)nullptr, 0, BSB + n * 16384, 128, tot);
    tot = 32 * 8 * 32;
    k_pack_B2<<<(tot + 255) / 256, 256, 0, stream>>>(sub_spline_w + n * 131072, 1024, 128,
                                                     (const float*)nullptr, 0, BSS + n * 131072, 128, tot);
  }
  { int tot = 12 * 32 * 32;
    k_pack_B2<<<(tot + 255) / 256, 256, 0, stream>>>(agg_w, 384, 512, (const float*)nullptr, 0, BAGG, 512, tot); }
  { int tot = 64 * 20 * 32;
    k_pack_A2<<<(tot + 255) / 256, 256, 0, stream>>>(inputs, 128, 128, h0, 512, AZ, 20, tot); }
  for (int n = 0; n < 3; ++n) {
    int tot = 64 * 8 * 32;
    k_pack_A2<<<(tot + 255) / 256, 256, 0, stream>>>(inputs, 128, 128, sub_states + n * 131072, 128,
                                                     ASUB + n * 262144, 8, tot);
  }

  // big fused signature-KAN GEMM -> H1 (12-way K-split, 32-wave WGs)
  k_kan_gemm<true, 8><<<dim3(8, 12, 1), 1024, 0, stream>>>(
      inputs, 128, SIGST, g_k1_ln_g, g_k1_ln_b, WB1, WS1, H1,
      16512, 43, 512, 0, 0, 0, 0, 0, 0);

  // LSTM gate pre-activations
  k_gemm_frag<<<dim3(8, 12), 256, 0, stream>>>(AZ, BZ, ZPRE, 20, 1536);

  // sub-layer recurrent aggregation
  for (int n = 0; n < 3; ++n)
    k_gemm_frag<<<dim3(8, 1), 256, 0, stream>>>(ASUB + n * 262144, BSRK + n * 32768,
                                                AGGIN + n * 131072, 8, 128);
  k_stats_rows<<<768, 128, 0, stream>>>(AGGIN, SUBST, 128, 3072);

  // sub KANLinear GEMMs (z = sub index)
  k_kan_gemm<false, 2><<<dim3(8, 1, 3), 256, 0, stream>>>(
      AGGIN, 128, SUBST, sub_ln_g, sub_ln_b, BSB, BSS, SUBKAN,
      128, 4, 128, 131072, 1024, 128, 16384, 131072, 131072);

  // small epilogue path
  k_h2<<<128, 256, 0, stream>>>(H1, g_k2_ln_g, g_k2_ln_b, g_k2_base, g_k2_spline, H2v);
  k_skip<<<1024, 128, 0, stream>>>(inputs, g_skip_w, g_skip_b, SKIP);
  k_sigw<<<4, 256, 0, stream>>>(H2v, SKIP, g_gate_w, g_gate_b, g_sig_w, g_sig_b, g_ln_g, g_ln_b, SIGW);

  float* out = (float*)d_out;
  k_sub_final<<<1536, 256, 0, stream>>>(SUBKAN, SIGW, sub_kernel, sub_states, out + 1048576, AAGG);
  k_gemm_frag<<<dim3(8, 4), 256, 0, stream>>>(AAGG, BAGG, XOPRE, 12, 512);
  k_final<<<2048, 256, 0, stream>>>(ZPRE, bias, XOPRE, agg_b, c0, out);
}
